// MODEL_6828998001205
// MI455X (gfx1250) — matrix operand order repaired and checked by reading, not compiled, not run on hardware
//
#include <hip/hip_runtime.h>
#include <hip/hip_bf16.h>

#define B_  64
#define S_  512
#define E_  256
#define H_  1024
#define NQ_ 2000
#define G4H (4*H_)

typedef __attribute__((ext_vector_type(16))) _Float16 v16h;
typedef __attribute__((ext_vector_type(8)))  _Float16 v8h;
typedef __attribute__((ext_vector_type(8)))  float    v8f;

// ---------------- utility kernels ----------------
__global__ void cast_f32_f16(const float* __restrict__ s, _Float16* __restrict__ d, size_t n) {
  size_t i = (size_t)blockIdx.x * blockDim.x + threadIdx.x;
  size_t st = (size_t)gridDim.x * blockDim.x;
  for (; i < n; i += st) d[i] = (_Float16)s[i];
}

__global__ void bias_sum(const float* __restrict__ a, const float* __restrict__ b,
                         float* __restrict__ o, int n) {
  int i = blockIdx.x * blockDim.x + threadIdx.x;
  if (i < n) o[i] = a[i] + b[i];
}

__global__ void init_state(const float* __restrict__ ih, const float* __restrict__ ic,
                           _Float16* __restrict__ h0, _Float16* __restrict__ h1,
                           float* __restrict__ c0, float* __restrict__ c1) {
  int i = blockIdx.x * blockDim.x + threadIdx.x;
  if (i < B_*H_) {
    h0[i] = (_Float16)ih[i];
    h1[i] = (_Float16)ih[B_*H_ + i];
    c0[i] = ic[i];
    c1[i] = ic[B_*H_ + i];
  }
}

__global__ void zero2(float* a) { if (threadIdx.x < 2) a[threadIdx.x] = 0.f; }

// one block per (b,s); 256 threads cover E=256; output layout [S][B][E]
__global__ void embed_gather(const int* __restrict__ xd, const float* __restrict__ emb,
                             _Float16* __restrict__ out) {
  int rid = blockIdx.x;            // rid = b*S + s
  int b = rid >> 9, s = rid & 511;
  int id = xd[rid];
  const float* src = emb + (size_t)id * E_;
  out[(((size_t)s * B_) + b) * E_ + threadIdx.x] = (_Float16)src[threadIdx.x];
}

// ---------------- LSTM step (WMMA, double-buffered async->LDS) ----------------
// LDS map: buf0 A[4K] B[4K] | buf1 A[4K] B[4K] | gates [4][64][16] f32 = 16K
#define LDS_BUF_STRIDE 8192u
#define LDS_B_OFF      4096u
#define LDS_G_OFF      16384u
#define LDS_BYTES      32768

__device__ __forceinline__ v16h cat16(v8h lo, v8h hi) {
  return __builtin_shufflevector(lo, hi, 0,1,2,3,4,5,6,7,8,9,10,11,12,13,14,15);
}

__device__ __forceinline__ void async_ld16(unsigned lds_off, const _Float16* g) {
  // CDNA5 async copy global -> LDS (tracked with ASYNCcnt)
  asm volatile("global_load_async_to_lds_b128 %0, %1, off"
               :: "v"(lds_off), "v"(g) : "memory");
}

// stage one K-chunk: A tile [64 x 32] halfs + B tiles [4][16 n][32 k] halfs
__device__ __forceinline__ void issue_chunk(const _Float16* __restrict__ A,
                                            const _Float16* __restrict__ W,
                                            int K, int n0, int k0, unsigned sel) {
  unsigned t  = threadIdx.x;
  unsigned ar = t >> 2, as = t & 3;                      // A: 64 rows x 4 16B segs
  unsigned bg = t >> 6, bn = (t >> 2) & 15, bs = t & 3;  // B: 4 gates x 16 n x 4 segs
  unsigned base = sel * LDS_BUF_STRIDE;
  async_ld16(base + ar*64u + as*16u,
             A + (size_t)ar*K + k0 + as*8);
  async_ld16(base + LDS_B_OFF + ((bg*16u + bn)*32u + bs*8u)*2u,
             W + (size_t)(bg*(unsigned)H_ + (unsigned)n0 + bn)*K + k0 + bs*8);
}

__device__ __forceinline__ void compute_chunk(const unsigned char* smem, unsigned sel,
                                              int gate, int mhalf, unsigned lane,
                                              v8f& acc0, v8f& acc1) {
  const _Float16* ldsA = (const _Float16*)(smem + sel * LDS_BUF_STRIDE);
  const _Float16* ldsB = (const _Float16*)(smem + sel * LDS_BUF_STRIDE + LDS_B_OFF);

  // A fragments (16-bit A layout: lanes<16 K{0..7,16..23}, lanes>=16 K{8..15,24..31})
  unsigned mlo   = lane & 15u;
  unsigned koffA = (lane >> 4) << 3;
  const _Float16* pa0 = ldsA + ((unsigned)mhalf*32u +       mlo)*32u + koffA;
  const _Float16* pa1 = ldsA + ((unsigned)mhalf*32u + 16u + mlo)*32u + koffA;
  v16h a0 = cat16(*(const v8h*)pa0, *(const v8h*)(pa0 + 16));
  v16h a1 = cat16(*(const v8h*)pa1, *(const v8h*)(pa1 + 16));
  // B fragment (lane = N column; lanes 0-15 hold K{0..7,16..23}, lanes 16-31 K{8..15,24..31})
  const _Float16* pb = ldsB + ((unsigned)gate*16u + (lane & 15u))*32u + ((lane >> 4) << 3);
  v16h bf = cat16(*(const v8h*)pb, *(const v8h*)(pb + 16));

  acc0 = __builtin_amdgcn_wmma_f32_16x16x32_f16(false, a0, false, bf, (short)0, acc0, false, false);
  acc1 = __builtin_amdgcn_wmma_f32_16x16x32_f16(false, a1, false, bf, (short)0, acc1, false, false);
}

// gates[g, m, n0+n] += A[m, :] . W[g*H + n0 + n, :]   (W row-major [4H, K])
// Steady-state loop is branch-free: issue next chunk, partial-wait (<=2 outstanding
// means previous chunk landed, ASYNCcnt retires in order), barrier, 2x WMMA, barrier.
__device__ __forceinline__ void gemm_acc(const _Float16* __restrict__ A,
                                         const _Float16* __restrict__ W,
                                         int K, int n0, int gate, int mhalf,
                                         unsigned lane, unsigned char* smem,
                                         v8f& acc0, v8f& acc1) {
  const int nchunks = K >> 5;
  issue_chunk(A, W, K, n0, 0, 0u);
  for (int i = 0; i < nchunks - 1; ++i) {
    unsigned sel = (unsigned)i & 1u;
    issue_chunk(A, W, K, n0, (i + 1) << 5, sel ^ 1u);
    asm volatile("s_wait_asynccnt 2" ::: "memory");
    __syncthreads();
    compute_chunk(smem, sel, gate, mhalf, lane, acc0, acc1);
    __syncthreads();   // buf[sel] overwritten by prefetch two chunks ahead
  }
  asm volatile("s_wait_asynccnt 0" ::: "memory");
  __syncthreads();
  compute_chunk(smem, (unsigned)(nchunks - 1) & 1u, gate, mhalf, lane, acc0, acc1);
  __syncthreads();
}

__global__ __launch_bounds__(256) void lstm_step(
    const _Float16* __restrict__ Xin, int K1,
    const _Float16* __restrict__ Wih,
    const _Float16* __restrict__ Hprev,
    const _Float16* __restrict__ Whh,
    const float* __restrict__ bias,
    float* __restrict__ Cst,
    _Float16* __restrict__ Hout,
    _Float16* __restrict__ Hstore) {
  __shared__ __align__(16) unsigned char smem[LDS_BYTES];
  float* ldsG = (float*)&smem[LDS_G_OFF];

  const unsigned lane = threadIdx.x & 31u;
  const unsigned wave = threadIdx.x >> 5;
  const int gate  = (int)(wave >> 1);   // 0..3 -> i,f,g,o
  const int mhalf = (int)(wave & 1u);   // batch rows 0-31 / 32-63
  const int n0    = blockIdx.x * 16;    // 16 hidden units per block

  __builtin_prefetch(bias + n0, 0, 3);  // global_prefetch_b8 (L2-resident bias)

  v8f acc0 = {}; v8f acc1 = {};
  gemm_acc(Xin,   Wih, K1, n0, gate, mhalf, lane, smem, acc0, acc1);
  gemm_acc(Hprev, Whh, H_, n0, gate, mhalf, lane, smem, acc0, acc1);

  // C/D layout: VGPR j -> M = j (lanes 0-15) / j+8 (lanes 16-31); N = lane%16
  {
    unsigned mb = (unsigned)mhalf*32u + ((lane >> 4) << 3);
    unsigned nn = lane & 15u;
#pragma unroll
    for (int j = 0; j < 8; ++j) {
      ldsG[((unsigned)gate*64u + mb + (unsigned)j)*16u + nn]        = acc0[j];
      ldsG[((unsigned)gate*64u + mb + 16u + (unsigned)j)*16u + nn]  = acc1[j];
    }
  }
  __syncthreads();

  // fused LSTM cell: 64 batch x 16 hidden = 1024 elements, 4 per thread
#pragma unroll
  for (int r = 0; r < 4; ++r) {
    unsigned e = threadIdx.x + 256u * (unsigned)r;
    unsigned b = e >> 4, nn = e & 15u;
    unsigned col = (unsigned)n0 + nn;
    float iv = ldsG[(0u*64u + b)*16u + nn] + bias[0*H_ + col];
    float fv = ldsG[(1u*64u + b)*16u + nn] + bias[1*H_ + col];
    float gv = ldsG[(2u*64u + b)*16u + nn] + bias[2*H_ + col];
    float ov = ldsG[(3u*64u + b)*16u + nn] + bias[3*H_ + col];
    float si = 1.f / (1.f + __expf(-iv));
    float sf = 1.f / (1.f + __expf(-fv));
    float so = 1.f / (1.f + __expf(-ov));
    float cold = Cst[b*H_ + col];
    float cnew = sf * cold + si * tanhf(gv);
    float h    = so * tanhf(cnew);
    Cst[b*H_ + col]  = cnew;
    Hout[b*H_ + col] = (_Float16)h;
    if (Hstore) Hstore[b*H_ + col] = (_Float16)h;
  }
}

// ---------------- head: gathered GEMV + BCE ----------------
__global__ __launch_bounds__(256) void pred_bce(
    const _Float16* __restrict__ H2,    // [S][B][H]
    const _Float16* __restrict__ Wlin,  // [NQ][H] fp16
    const float* __restrict__ blin,
    const int* __restrict__ qt,
    const float* __restrict__ tgt,
    float* __restrict__ out_sig,
    float* __restrict__ out_tgt,
    float* __restrict__ acc) {
  int row  = blockIdx.x * 8 + (int)(threadIdx.x >> 5);  // row = b*S + s
  int lane = (int)(threadIdx.x & 31u);
  int b = row >> 9, s = row & 511;
  int q = qt[row] - 1;
  q = q < 0 ? 0 : (q > NQ_ - 1 ? NQ_ - 1 : q);
  const _Float16* hr = H2 + (((size_t)s * B_) + b) * H_;
  const _Float16* wr = Wlin + (size_t)q * H_;
  __builtin_prefetch(wr + lane * 32, 0, 0);  // pull gathered Wlin row toward L2/L0
  float sum = 0.f;
#pragma unroll
  for (int it = 0; it < 4; ++it) {
    int k = it * 256 + lane * 8;
    v8h hv = *(const v8h*)(hr + k);
    v8h wv = *(const v8h*)(wr + k);
#pragma unroll
    for (int j = 0; j < 8; ++j) sum += (float)hv[j] * (float)wv[j];
  }
#pragma unroll
  for (int off = 16; off > 0; off >>= 1) sum += __shfl_xor(sum, off, 32);
  if (lane == 0) {
    float x  = sum + blin[q];
    float sg = 1.f / (1.f + __expf(-x));
    float t  = tgt[row];
    out_sig[row] = sg;
    out_tgt[row] = t;
    if (t >= 0.f) {
      float bce = fmaxf(x, 0.f) - x * t + log1pf(__expf(-fabsf(x)));
      atomicAdd(&acc[0], bce);
      atomicAdd(&acc[1], 1.f);
    }
  }
}

__global__ void finalize(const float* __restrict__ acc, float* __restrict__ out) {
  if (threadIdx.x == 0) out[0] = acc[0] / acc[1];
}

// ---------------- host ----------------
extern "C" void kernel_launch(void* const* d_in, const int* in_sizes, int n_in,
                              void* d_out, int out_size, void* d_ws, size_t ws_size,
                              hipStream_t stream) {
  (void)in_sizes; (void)n_in; (void)out_size; (void)ws_size;
  const int*   x_data  = (const int*)  d_in[0];
  const int*   q_t     = (const int*)  d_in[1];
  const float* target  = (const float*)d_in[2];
  const float* x_embed = (const float*)d_in[3];
  const float* Wih0    = (const float*)d_in[4];
  const float* Whh0    = (const float*)d_in[5];
  const float* bih0    = (const float*)d_in[6];
  const float* bhh0    = (const float*)d_in[7];
  const float* Wih1    = (const float*)d_in[8];
  const float* Whh1    = (const float*)d_in[9];
  const float* bih1    = (const float*)d_in[10];
  const float* bhh1    = (const float*)d_in[11];
  const float* Wlin    = (const float*)d_in[12];
  const float* blin    = (const float*)d_in[13];
  const float* init_h  = (const float*)d_in[14];
  const float* init_c  = (const float*)d_in[15];

  char* ws = (char*)d_ws;
  size_t off = 0;
  auto take = [&](size_t bytes) -> char* {
    char* p = ws + off;
    off = (off + bytes + 255) & ~(size_t)255;
    return p;
  };
  _Float16* Wih0h = (_Float16*)take((size_t)G4H * E_ * 2);
  _Float16* Whh0h = (_Float16*)take((size_t)G4H * H_ * 2);
  _Float16* Wih1h = (_Float16*)take((size_t)G4H * H_ * 2);
  _Float16* Whh1h = (_Float16*)take((size_t)G4H * H_ * 2);
  _Float16* Wlinh = (_Float16*)take((size_t)NQ_ * H_ * 2);
  float*    bias0 = (float*)take((size_t)G4H * 4);
  float*    bias1 = (float*)take((size_t)G4H * 4);
  _Float16* embT  = (_Float16*)take((size_t)S_ * B_ * E_ * 2);
  _Float16* h0A   = (_Float16*)take((size_t)B_ * H_ * 2);
  _Float16* h0B   = (_Float16*)take((size_t)B_ * H_ * 2);
  _Float16* h1A   = (_Float16*)take((size_t)B_ * H_ * 2);
  _Float16* h1B   = (_Float16*)take((size_t)B_ * H_ * 2);
  float*    c0    = (float*)take((size_t)B_ * H_ * 4);
  float*    c1    = (float*)take((size_t)B_ * H_ * 4);
  _Float16* h2    = (_Float16*)take((size_t)S_ * B_ * H_ * 2);
  float*    acc   = (float*)take(256);

  cast_f32_f16<<<1024, 256, 0, stream>>>(Wih0, Wih0h, (size_t)G4H * E_);
  cast_f32_f16<<<2048, 256, 0, stream>>>(Whh0, Whh0h, (size_t)G4H * H_);
  cast_f32_f16<<<2048, 256, 0, stream>>>(Wih1, Wih1h, (size_t)G4H * H_);
  cast_f32_f16<<<2048, 256, 0, stream>>>(Whh1, Whh1h, (size_t)G4H * H_);
  cast_f32_f16<<<2048, 256, 0, stream>>>(Wlin, Wlinh, (size_t)NQ_ * H_);
  bias_sum<<<(G4H + 255) / 256, 256, 0, stream>>>(bih0, bhh0, bias0, G4H);
  bias_sum<<<(G4H + 255) / 256, 256, 0, stream>>>(bih1, bhh1, bias1, G4H);
  init_state<<<(B_ * H_ + 255) / 256, 256, 0, stream>>>(init_h, init_c, h0A, h1A, c0, c1);
  zero2<<<1, 32, 0, stream>>>(acc);
  embed_gather<<<B_ * S_, 256, 0, stream>>>(x_data, x_embed, embT);

  _Float16 *h0p = h0A, *h0n = h0B, *h1p = h1A, *h1n = h1B;
  for (int t = 0; t < S_; ++t) {
    lstm_step<<<64, 256, 0, stream>>>(embT + (size_t)t * B_ * E_, E_, Wih0h,
                                      h0p, Whh0h, bias0, c0, h0n, (_Float16*)nullptr);
    lstm_step<<<64, 256, 0, stream>>>(h0n, H_, Wih1h,
                                      h1p, Whh1h, bias1, c1, h1n, h2 + (size_t)t * B_ * H_);
    { _Float16* tmp = h0p; h0p = h0n; h0n = tmp; }
    { _Float16* tmp = h1p; h1p = h1n; h1n = tmp; }
  }

  float* out = (float*)d_out;
  pred_bce<<<(B_ * S_) / 8, 256, 0, stream>>>(h2, Wlinh, blin, q_t, target,
                                              out + 1, out + 1 + B_ * S_, acc);
  finalize<<<1, 32, 0, stream>>>(acc, out);
}
